// LinearAttention_87565793230944
// MI455X (gfx1250) — matrix operand order repaired and checked by reading, not compiled, not run on hardware
//
#include <hip/hip_runtime.h>
#include <stdint.h>

typedef __attribute__((ext_vector_type(16))) __bf16 bf16x16;
typedef __attribute__((ext_vector_type(8)))  float  f32x8;
typedef int v4i_g __attribute__((vector_size(16)));   // matches builtin param pointee

constexpr int kDModel  = 1024;
constexpr int kHeads   = 16;
constexpr int kHeadDim = 64;
constexpr int kSeq     = 2048;
constexpr int kBatch   = 2;
constexpr int kM       = kBatch * kSeq;   // 4096
constexpr float kEps   = 1e-6f;

constexpr int BM = 128, BN = 128, BK = 32;
constexpr int LDSS = 40;                  // ushort stride: 80B rows, 16B aligned, conflict-free
constexpr int TILE_US = BM * LDSS;        // ushorts per tile buffer

// Async global->LDS path (device pass only; host pass parses the fallback).
#if defined(__has_builtin)
#  if __has_builtin(__builtin_amdgcn_global_load_async_to_lds_b128)
#    define USE_ASYNC_LDS 1
#  endif
#endif
#ifndef USE_ASYNC_LDS
#  define USE_ASYNC_LDS 0
#endif

__device__ __forceinline__ unsigned short f2bf(float f) {
  union { float f; uint32_t u; } c; c.f = f;
  uint32_t u = c.u;
  u += 0x7FFFu + ((u >> 16) & 1u);        // round-to-nearest-even
  return (unsigned short)(u >> 16);
}
__device__ __forceinline__ float bf2f(uint32_t h) {
  union { uint32_t u; float f; } c; c.u = h << 16;
  return c.f;
}

// ---------------------------------------------------------------------------
// fp32 -> bf16 pack (one-shot; all later kernels consume bf16)
// ---------------------------------------------------------------------------
__global__ __launch_bounds__(256)
void cvt_f32_to_bf16(const float* __restrict__ in, unsigned short* __restrict__ out, int n4) {
  int i = blockIdx.x * 256 + threadIdx.x;
  if (i >= n4) return;
  float4 f = ((const float4*)in)[i];
  uint2 o;
  o.x = (uint32_t)f2bf(f.x) | ((uint32_t)f2bf(f.y) << 16);
  o.y = (uint32_t)f2bf(f.z) | ((uint32_t)f2bf(f.w) << 16);
  ((uint2*)out)[i] = o;
}

// ---------------------------------------------------------------------------
// GEMM: Out[M,N] = A[M,K] @ B[N,K]^T, bf16 operands, f32 WMMA accumulate.
// Double-buffered LDS, async global->LDS staging when available.
// MODE 0: f32 row-major store (final projection -> d_out)
// MODE 1: relu+eps, bf16 scatter to (b,h,s,d)    (q, k)
// MODE 2: bf16 scatter to (b,h,s,d)              (v)
// ---------------------------------------------------------------------------
template <int MODE>
__global__ __launch_bounds__(256)
void gemm_bf16_wmma(const unsigned short* __restrict__ A,
                    const unsigned short* __restrict__ B,
                    void* __restrict__ Out,
                    int K, int Ntot) {
  __shared__ unsigned short ldsA[2 * TILE_US];
  __shared__ unsigned short ldsB[2 * TILE_US];

  const int tid  = threadIdx.x;
  const int lane = tid & 31;
  const int wid  = tid >> 5;
  const int wm   = wid >> 2;            // 0..1 -> 64 rows each
  const int wn   = wid & 3;             // 0..3 -> 32 cols each
  const int m0   = blockIdx.y * BM;
  const int n0   = blockIdx.x * BN;
  const int hi   = (lane >> 4) & 1;
  const int l15  = lane & 15;

  f32x8 acc[4][2] = {};

  auto stage = [&](int buf, int kt) {
    #pragma unroll
    for (int it = 0; it < 2; ++it) {
      int fid = tid + it * 256;             // 0..511
      int row = fid >> 2;                   // 0..127
      int c8  = (fid & 3) * 8;              // 0,8,16,24 (ushorts)
      const unsigned short* ga = A + (size_t)(m0 + row) * K + kt + c8;
      const unsigned short* gb = B + (size_t)(n0 + row) * K + kt + c8;
      unsigned short* la = &ldsA[buf * TILE_US + row * LDSS + c8];
      unsigned short* lb = &ldsB[buf * TILE_US + row * LDSS + c8];
#if USE_ASYNC_LDS
      __builtin_amdgcn_global_load_async_to_lds_b128(
          (__attribute__((address_space(1))) v4i_g*)ga,
          (__attribute__((address_space(3))) v4i_g*)la, 0, 0);
      __builtin_amdgcn_global_load_async_to_lds_b128(
          (__attribute__((address_space(1))) v4i_g*)gb,
          (__attribute__((address_space(3))) v4i_g*)lb, 0, 0);
#else
      uint4 ua = *(const uint4*)ga;
      uint4 ub = *(const uint4*)gb;
      uint32_t* pa = (uint32_t*)la; pa[0] = ua.x; pa[1] = ua.y; pa[2] = ua.z; pa[3] = ua.w;
      uint32_t* pb = (uint32_t*)lb; pb[0] = ub.x; pb[1] = ub.y; pb[2] = ub.z; pb[3] = ub.w;
#endif
    }
  };
  auto wait_async = [&]() {
#if USE_ASYNC_LDS
    asm volatile("s_wait_asynccnt 0x0" ::: "memory");
#endif
  };

  int buf = 0;
  stage(0, 0);
  wait_async();
  __syncthreads();

  for (int kt = 0; kt < K; kt += BK) {
    if (kt + BK < K) stage(buf ^ 1, kt + BK);   // overlap with WMMA below

    // ---- B fragments: lane = N; hi half holds K+8; 16B chunks at K and K+16 ----
    bf16x16 bfrag[2];
    #pragma unroll
    for (int j = 0; j < 2; ++j) {
      union { bf16x16 v; uint4 q[2]; } r;
      const uint4* p4 = (const uint4*)(ldsB + buf * TILE_US +
                                       (wn * 32 + j * 16 + l15) * LDSS + hi * 8);
      r.q[0] = p4[0];
      r.q[1] = p4[2];
      bfrag[j] = r.v;
    }
    // ---- A fragments (ISA pair layout) + WMMA ----
    #pragma unroll
    for (int i = 0; i < 4; ++i) {
      union { bf16x16 v; uint32_t u[8]; } r;
      const unsigned short* p = ldsA + buf * TILE_US + (wm * 64 + i * 16 + l15) * LDSS;
      #pragma unroll
      for (int vv = 0; vv < 8; ++vv) {
        int kb = ((vv < 4) ? 0 : 16) + hi * 8 + (vv & 3) * 2;
        r.u[vv] = *(const uint32_t*)(p + kb);
      }
      #pragma unroll
      for (int j = 0; j < 2; ++j)
        acc[i][j] = __builtin_amdgcn_wmma_f32_16x16x32_bf16(
            false, r.v, false, bfrag[j], (short)0, acc[i][j], false, false);
    }
    wait_async();
    __syncthreads();
    buf ^= 1;
  }

  // ---- epilogue: C/D layout VGPR r -> M = r + 8*hi, N = lane&15 ----
  #pragma unroll
  for (int i = 0; i < 4; ++i) {
    #pragma unroll
    for (int j = 0; j < 2; ++j) {
      #pragma unroll
      for (int r = 0; r < 8; ++r) {
        int Mg = m0 + wm * 64 + i * 16 + r + hi * 8;
        int Ng = n0 + wn * 32 + j * 16 + l15;
        float val = acc[i][j][r];
        if (MODE == 0) {
          ((float*)Out)[(size_t)Mg * Ntot + Ng] = val;
        } else {
          int b = Mg >> 11, s = Mg & (kSeq - 1);
          int h = Ng >> 6,  d = Ng & (kHeadDim - 1);
          size_t idx = (((size_t)b * kHeads + h) * kSeq + s) * kHeadDim + d;
          if (MODE == 1) val = fmaxf(val, 0.0f) + kEps;
          ((unsigned short*)Out)[idx] = f2bf(val);
        }
      }
    }
  }
}

// ---------------------------------------------------------------------------
// Causal linear-attention scan. One block per (b,h); 64 threads; thread m owns
// output column m (kv column + redundant k-cumsum in regs). bf16 in/out.
// ---------------------------------------------------------------------------
__device__ __forceinline__ void st8(float* d, uint4 u) {
  d[0] = bf2f(u.x & 0xffffu); d[1] = bf2f(u.x >> 16);
  d[2] = bf2f(u.y & 0xffffu); d[3] = bf2f(u.y >> 16);
  d[4] = bf2f(u.z & 0xffffu); d[5] = bf2f(u.z >> 16);
  d[6] = bf2f(u.w & 0xffffu); d[7] = bf2f(u.w >> 16);
}

__global__ __launch_bounds__(64)
void linattn_scan(const unsigned short* __restrict__ Q,
                  const unsigned short* __restrict__ Kf,
                  const unsigned short* __restrict__ V,
                  unsigned short* __restrict__ Yb) {
  __shared__ float sq[64 * 64];
  __shared__ float sk[64 * 64];
  __shared__ float sv[64 * 64];

  const int bh = blockIdx.x;
  const int b  = bh >> 4;
  const int h  = bh & 15;
  const int m  = threadIdx.x;

  float kvc[64], kc[64];
  #pragma unroll
  for (int d = 0; d < 64; ++d) { kvc[d] = 0.f; kc[d] = 0.f; }

  const size_t base = (size_t)bh * kSeq * kHeadDim;

  for (int chunk = 0; chunk < kSeq / 64; ++chunk) {
    const size_t cb = base + (size_t)chunk * 64 * kHeadDim;
    for (int t = m; t < 64 * 64 / 8; t += 64) {      // 8 bf16 per uint4
      uint4 uq = ((const uint4*)(Q  + cb))[t];
      uint4 uk = ((const uint4*)(Kf + cb))[t];
      uint4 uv = ((const uint4*)(V  + cb))[t];
      st8(sq + t * 8, uq);
      st8(sk + t * 8, uk);
      st8(sv + t * 8, uv);
    }
    __syncthreads();

    #pragma unroll 1
    for (int i = 0; i < 64; ++i) {
      const float vm = sv[i * 64 + m];
      float o[4] = {0.f, 0.f, 0.f, 0.f};
      float z[4] = {0.f, 0.f, 0.f, 0.f};
      #pragma unroll
      for (int d = 0; d < 64; ++d) {
        float kd = sk[i * 64 + d];
        float qd = sq[i * 64 + d];
        kc[d]  += kd;                          // inclusive k cumsum
        kvc[d]  = fmaf(kd, vm, kvc[d]);        // inclusive kv cumsum column
        o[d & 3] = fmaf(qd, kvc[d], o[d & 3]);
        z[d & 3] = fmaf(qd, kc[d],  z[d & 3]);
      }
      float on = (o[0] + o[1]) + (o[2] + o[3]);
      float zn = (z[0] + z[1]) + (z[2] + z[3]);
      float r  = on / (zn + kEps);
      int s = chunk * 64 + i;
      Yb[((size_t)b * kSeq + s) * kDModel + h * kHeadDim + m] = f2bf(r);
    }
    __syncthreads();
  }
}

// ---------------------------------------------------------------------------
extern "C" void kernel_launch(void* const* d_in, const int* in_sizes, int n_in,
                              void* d_out, int out_size, void* d_ws, size_t ws_size,
                              hipStream_t stream) {
  (void)in_sizes; (void)n_in; (void)out_size; (void)ws_size;
  const float* x  = (const float*)d_in[0];
  const float* Wq = (const float*)d_in[1];
  const float* Wk = (const float*)d_in[2];
  const float* Wv = (const float*)d_in[3];
  const float* Wo = (const float*)d_in[4];

  const size_t nX = (size_t)kM * kDModel;        // 4M elements
  const size_t nW = (size_t)kDModel * kDModel;   // 1M elements

  unsigned short* xb  = (unsigned short*)d_ws;   // bf16 x        (8 MB)
  unsigned short* wqb = xb  + nX;                // bf16 weights  (2 MB each)
  unsigned short* wkb = wqb + nW;
  unsigned short* wvb = wkb + nW;
  unsigned short* wob = wvb + nW;
  unsigned short* qb  = wob + nW;                // bf16 q (b,h,s,d) (8 MB each)
  unsigned short* kb  = qb  + nX;
  unsigned short* vb  = kb  + nX;
  unsigned short* yb  = vb  + nX;                // bf16 y (b,s,dmodel)

  // one-shot conversions
  hipLaunchKernelGGL(cvt_f32_to_bf16, dim3((int)(nX / 4 / 256)), dim3(256), 0, stream, x,  xb,  (int)(nX / 4));
  hipLaunchKernelGGL(cvt_f32_to_bf16, dim3((int)(nW / 4 / 256)), dim3(256), 0, stream, Wq, wqb, (int)(nW / 4));
  hipLaunchKernelGGL(cvt_f32_to_bf16, dim3((int)(nW / 4 / 256)), dim3(256), 0, stream, Wk, wkb, (int)(nW / 4));
  hipLaunchKernelGGL(cvt_f32_to_bf16, dim3((int)(nW / 4 / 256)), dim3(256), 0, stream, Wv, wvb, (int)(nW / 4));
  hipLaunchKernelGGL(cvt_f32_to_bf16, dim3((int)(nW / 4 / 256)), dim3(256), 0, stream, Wo, wob, (int)(nW / 4));

  dim3 grid(kDModel / BN, kM / BM);
  dim3 blk(256);
  hipLaunchKernelGGL((gemm_bf16_wmma<1>), grid, blk, 0, stream, xb, wqb, (void*)qb, kDModel, kDModel);
  hipLaunchKernelGGL((gemm_bf16_wmma<1>), grid, blk, 0, stream, xb, wkb, (void*)kb, kDModel, kDModel);
  hipLaunchKernelGGL((gemm_bf16_wmma<2>), grid, blk, 0, stream, xb, wvb, (void*)vb, kDModel, kDModel);

  hipLaunchKernelGGL(linattn_scan, dim3(kBatch * kHeads), dim3(64), 0, stream, qb, kb, vb, yb);

  hipLaunchKernelGGL((gemm_bf16_wmma<0>), grid, blk, 0, stream, yb, wob, d_out, kDModel, kDModel);
}
